// GraphAttentionLayer_40278203301986
// MI455X (gfx1250) — compile-verified, matrix operand order checked by reading, not run on hardware
//
#include <hip/hip_runtime.h>
#include <hip/hip_bf16.h>
#include <math.h>

// GAT layer, algebraically refactored:
//   attention logits computed directly from raw features via precomputed
//   q = W_c@(wa_lo+wa_hi), p = W_d@wa_lo, r = W_d@wa_hi
//   out = elu( (sum_k attn_k * nb_k) @ W_d + b_d )
// Phase B streams the 512MiB neighbor tensor once (HBM-bound, ~24us floor
// at 23.3 TB/s); phase C is a 32768x128x128 GEMM on v_wmma_f32_16x16x32_bf16
// with hi/lo bf16 splitting (3 WMMAs / k-step) for near-fp32 accuracy.

typedef __attribute__((ext_vector_type(16))) __bf16 v16bf;
typedef __attribute__((ext_vector_type(8)))  float  v8f;

#define GAT_ALPHA 0.2f
static constexpr int N_NODES = 32768;
static constexpr int KNBR    = 32;
static constexpr int DIM     = 128;

__device__ __forceinline__ float wave_sum(float v) {
#pragma unroll
  for (int o = 16; o > 0; o >>= 1) v += __shfl_xor(v, o, 32);
  return v;
}
__device__ __forceinline__ float lrelu(float x) { return x >= 0.f ? x : GAT_ALPHA * x; }

__device__ __forceinline__ void split_bf16(float x, __bf16& hi, __bf16& lo) {
  __bf16 h = (__bf16)x;
  hi = h;
  lo = (__bf16)(x - (float)h);
}

// ---------------- Kernel A: tiny precompute of q, p, r, c_c, c_n ------------
__global__ void __launch_bounds__(128)
gat_precompute(const float* __restrict__ Wc, const float* __restrict__ bc,
               const float* __restrict__ Wd, const float* __restrict__ bd,
               const float* __restrict__ wa, const float* __restrict__ ba,
               float* __restrict__ ws) {
  const int f = threadIdx.x;  // 0..127
  float q = 0.f, p = 0.f, r = 0.f;
  for (int d = 0; d < DIM; ++d) {
    const float lo = wa[d], hi = wa[DIM + d];
    q = fmaf(Wc[f * DIM + d], lo + hi, q);
    p = fmaf(Wd[f * DIM + d], lo, p);
    r = fmaf(Wd[f * DIM + d], hi, r);
  }
  ws[f] = q;
  ws[DIM + f] = p;
  ws[2 * DIM + f] = r;
  if (f == 0) {
    float cc = ba[0], cn = ba[0];
    for (int d = 0; d < DIM; ++d) {
      const float s = wa[d] + wa[DIM + d];
      cc = fmaf(bc[d], s, cc);
      cn = fmaf(bd[d], s, cn);
    }
    ws[3 * DIM + 0] = cc;
    ws[3 * DIM + 1] = cn;
  }
}

// ------- Kernel B: stream neighbors once; softmax-weighted row sum ----------
// One wave per node; the node's 32x128 tile is register-resident.
__global__ void __launch_bounds__(256)
gat_attn(const float* __restrict__ combin, const float* __restrict__ nbrs,
         const float* __restrict__ ws, float* __restrict__ g_out) {
  const int lane = threadIdx.x & 31;
  const int node = blockIdx.x * 8 + (threadIdx.x >> 5);

  const float4* wsv = (const float4*)ws;
  const float4 q4 = wsv[lane];
  const float4 p4 = wsv[32 + lane];
  const float4 r4 = wsv[64 + lane];
  const float c_c = ws[3 * DIM + 0];
  const float c_n = ws[3 * DIM + 1];

  // Coalesced: each lane owns features [4*lane, 4*lane+4) of every row.
  const float4* nb = (const float4*)(nbrs + (size_t)node * (KNBR * DIM));
  float4 row[KNBR];
#pragma unroll
  for (int k = 0; k < KNBR; ++k) row[k] = nb[k * 32 + lane];

  const float4 c4 = ((const float4*)(combin + (size_t)node * DIM))[lane];
  const float sc = wave_sum(c4.x * q4.x + c4.y * q4.y + c4.z * q4.z + c4.w * q4.w) + c_c;
  const float lc = lrelu(sc);

  float lm[16];
  float mx = lc;
  float tprev = 0.f;
#pragma unroll
  for (int k = 0; k < KNBR; ++k) {
    const float4 pr = (k & 1) ? r4 : p4;
    const float4 a = row[k];
    const float t = wave_sum(a.x * pr.x + a.y * pr.y + a.z * pr.z + a.w * pr.w);
    if (k & 1) {
      const float l = lrelu(tprev + t + c_n);
      lm[k >> 1] = l;
      mx = fmaxf(mx, l);
    } else {
      tprev = t;
    }
  }

  float wc = __expf(lc - mx);
  float Z = 16.f * wc;  // 16 identical slots share the combin logit
  float wm[16];
#pragma unroll
  for (int m = 0; m < 16; ++m) {
    wm[m] = __expf(lm[m] - mx);
    Z += wm[m];
  }
  const float inv = 1.f / Z;
  wc *= inv;

  float4 g;
  g.x = g.y = g.z = g.w = 0.f;
#pragma unroll
  for (int k = 0; k < 16; ++k) {
    g.x = fmaf(wc, row[k].x, g.x);
    g.y = fmaf(wc, row[k].y, g.y);
    g.z = fmaf(wc, row[k].z, g.z);
    g.w = fmaf(wc, row[k].w, g.w);
  }
#pragma unroll
  for (int m = 0; m < 16; ++m) {
    const float w = wm[m] * inv;
    const float4 a = row[16 + m];
    g.x = fmaf(w, a.x, g.x);
    g.y = fmaf(w, a.y, g.y);
    g.z = fmaf(w, a.z, g.z);
    g.w = fmaf(w, a.w, g.w);
  }
  ((float4*)(g_out + (size_t)node * DIM))[lane] = g;
}

// ------- Kernel C: out = elu(g @ W_d + b_d), in place, WMMA bf16 hi/lo ------
// One wave per 16-row M-tile. A layout per CDNA5 ISA 16-bit A (16x32):
// lane half selects K block {0..7|16..23} vs {8..15|24..31}; VGPR v packs
// the (even,odd) K pair. B mirrors it over columns.
__global__ void __launch_bounds__(256)
gat_gemm_elu(const float* __restrict__ Wd, const float* __restrict__ bd,
             float* __restrict__ io) {
  const int lane = threadIdx.x & 31;
  const int tm   = blockIdx.x * 8 + (threadIdx.x >> 5);  // 0..2047
  const int ln   = lane & 15;
  const int half = lane >> 4;

  // Load the wave's 16 g-rows (all 128 cols) into bf16 hi/lo A fragments.
  v16bf ah[4], al[4];
#pragma unroll
  for (int s = 0; s < 4; ++s) {
#pragma unroll
    for (int v = 0; v < 8; ++v) {
      const int kk = (v < 4) ? (8 * half + 2 * v) : (16 + 8 * half + 2 * (v - 4));
      const float* rp = io + ((size_t)(tm * 16 + ln)) * DIM + s * 32 + kk;
      const float x0 = rp[0], x1 = rp[1];
      __bf16 h0, l0, h1, l1;
      split_bf16(x0, h0, l0);
      split_bf16(x1, h1, l1);
      ah[s][2 * v] = h0;  al[s][2 * v] = l0;
      ah[s][2 * v + 1] = h1;  al[s][2 * v + 1] = l1;
    }
  }

#pragma unroll
  for (int tn = 0; tn < 8; ++tn) {
    v8f acc = {};
#pragma unroll
    for (int s = 0; s < 4; ++s) {
      v16bf bh, bl;
#pragma unroll
      for (int v = 0; v < 8; ++v) {
        const int kk = (v < 4) ? (8 * half + 2 * v) : (16 + 8 * half + 2 * (v - 4));
        const int kr = s * 32 + kk;
        const int col = tn * 16 + ln;
        const float y0 = Wd[(size_t)kr * DIM + col];
        const float y1 = Wd[(size_t)(kr + 1) * DIM + col];
        __bf16 h0, l0, h1, l1;
        split_bf16(y0, h0, l0);
        split_bf16(y1, h1, l1);
        bh[2 * v] = h0;  bl[2 * v] = l0;
        bh[2 * v + 1] = h1;  bl[2 * v + 1] = l1;
      }
      // (neg_a, A, neg_b, B, c_mod, C, reuse_a, reuse_b)
      acc = __builtin_amdgcn_wmma_f32_16x16x32_bf16(false, ah[s], false, bh,
                                                    (short)0, acc, false, false);
      acc = __builtin_amdgcn_wmma_f32_16x16x32_bf16(false, ah[s], false, bl,
                                                    (short)0, acc, false, false);
      acc = __builtin_amdgcn_wmma_f32_16x16x32_bf16(false, al[s], false, bh,
                                                    (short)0, acc, false, false);
    }
    const int col = tn * 16 + ln;
    const float bias = bd[col];
#pragma unroll
    for (int v = 0; v < 8; ++v) {
      float val = acc[v] + bias;
      val = (val > 0.f) ? val : (__expf(val) - 1.f);  // ELU(alpha=1)
      io[((size_t)(tm * 16 + v + 8 * half)) * DIM + col] = val;
    }
  }
}

extern "C" void kernel_launch(void* const* d_in, const int* in_sizes, int n_in,
                              void* d_out, int out_size, void* d_ws, size_t ws_size,
                              hipStream_t stream) {
  const float* combin = (const float*)d_in[0];  // [N, 128]
  const float* nbrs   = (const float*)d_in[1];  // [N, 32, 128]
  const float* Wc     = (const float*)d_in[2];  // [128, 128]
  const float* bc     = (const float*)d_in[3];  // [128]
  const float* Wd     = (const float*)d_in[4];  // [128, 128]
  const float* bd     = (const float*)d_in[5];  // [128]
  const float* wa     = (const float*)d_in[6];  // [256]
  const float* ba     = (const float*)d_in[7];  // [1]
  float* out = (float*)d_out;                   // [N, 128]; also holds g between B and C
  float* ws  = (float*)d_ws;                    // 386 floats used

  gat_precompute<<<1, 128, 0, stream>>>(Wc, bc, Wd, bd, wa, ba, ws);
  gat_attn<<<N_NODES / 8, 256, 0, stream>>>(combin, nbrs, ws, out);
  gat_gemm_elu<<<(N_NODES / 16) / 8, 256, 0, stream>>>(Wd, bd, out);
}
